// ProbSparseMHA_16879221473962
// MI455X (gfx1250) — matrix operand order repaired and checked by reading, not compiled, not run on hardware
//
#include <hip/hip_runtime.h>
#include <hip/hip_bf16.h>
#include <stdint.h>

#define B_SZ 2
#define T_SEQ 4096
#define D_MODEL 1024
#define H_CNT 16
#define HEAD_D 64
#define KEEP_N 1228
#define KEEP_PAD 1280           // next multiple of 64; tail zero-filled
#define M_TOT (B_SZ * T_SEQ)    // 8192
#define N_QKV (3 * D_MODEL)     // 3072

typedef __attribute__((ext_vector_type(16))) __bf16 v16bf;
typedef __attribute__((ext_vector_type(8)))  float  v8f;
typedef __attribute__((ext_vector_type(8)))  unsigned int v8u;

union FragU { v8u u; v16bf b; };
union Pack4 { uint4 q; unsigned short s[8]; };

static __device__ __forceinline__ unsigned short f2bf(float f) {
  unsigned int u = __float_as_uint(f);
  u += 0x7fffu + ((u >> 16) & 1u);   // round-to-nearest-even
  return (unsigned short)(u >> 16);
}
static __device__ __forceinline__ float bf2f(unsigned short h) {
  return __uint_as_float(((unsigned int)h) << 16);
}
// CDNA5 16-bit A-matrix 16x32 layout: VGPR p holds a K pair; lanes 16-31 shift K by 8.
static __device__ __forceinline__ int a_kmap(int p, int half) {
  return (p < 4 ? 2 * p : 8 + 2 * p) + half * 8;
}

// ---------------------------------------------------------------------------
// f32 -> bf16 conversion
// ---------------------------------------------------------------------------
__global__ void f32_to_bf16_kernel(const float* __restrict__ in,
                                   unsigned short* __restrict__ out, int n) {
  int i = blockIdx.x * blockDim.x + threadIdx.x;
  int stride = gridDim.x * blockDim.x;
  for (; i < n; i += stride) out[i] = f2bf(in[i]);
}

// ---------------------------------------------------------------------------
// Tiled bf16 WMMA GEMM: C[M,N] = A[M,K] * B[K,N] + bias
// MODE 0: scatter output as bf16 q/k/v in [B,H,T,d] layout
// MODE 1: f32 row-major output
// Block = 256 threads (8 waves). Tile 128(M) x 64(N), K-step 32.
// ---------------------------------------------------------------------------
template <int MODE>
__global__ __launch_bounds__(256) void gemm_bf16_wmma(
    const unsigned short* __restrict__ A,
    const unsigned short* __restrict__ Bm,
    const float* __restrict__ bias,
    float* __restrict__ Cf,
    unsigned short* __restrict__ Qo,
    unsigned short* __restrict__ Ko,
    unsigned short* __restrict__ Vo,
    int M, int N, int K) {
  __shared__ __align__(16) unsigned short As[128 * 32];  // [m][k]
  __shared__ __align__(16) unsigned short Bs[64 * 32];   // [n][k] (transposed)
  const int tid  = threadIdx.x;
  const int wave = tid >> 5;
  const int lane = tid & 31;
  const int l15  = lane & 15;
  const int half = lane >> 4;
  const int m0 = blockIdx.x * 128;
  const int n0 = blockIdx.y * 64;

  v8f acc[4];
  #pragma unroll
  for (int a = 0; a < 4; ++a)
    #pragma unroll
    for (int i = 0; i < 8; ++i) acc[a][i] = 0.0f;

  for (int k0 = 0; k0 < K; k0 += 32) {
    {   // A tile: each thread moves 16 bf16 (two uint4), row-major coalesced
      int row = tid >> 1;
      int kk  = (tid & 1) << 4;
      const uint4* src = reinterpret_cast<const uint4*>(
          A + (size_t)(m0 + row) * K + k0 + kk);
      uint4 p0 = src[0];
      uint4 p1 = src[1];
      uint4* dst = reinterpret_cast<uint4*>(&As[row * 32 + kk]);
      dst[0] = p0;
      dst[1] = p1;
      if (k0 + 32 < K)
        __builtin_prefetch(A + (size_t)(m0 + row) * K + k0 + 32, 0, 1);
    }
    {   // B tile: coalesced uint4 row reads, transposed scalar LDS writes
      int k  = tid >> 3;             // 0..31
      int nb = (tid & 7) << 3;       // 0,8,..,56
      Pack4 pk;
      pk.q = *reinterpret_cast<const uint4*>(Bm + (size_t)(k0 + k) * N + n0 + nb);
      #pragma unroll
      for (int j = 0; j < 8; ++j)
        Bs[(nb + j) * 32 + k] = pk.s[j];
    }
    __syncthreads();

    FragU af;
    #pragma unroll
    for (int p = 0; p < 8; ++p)
      af.u[p] = *reinterpret_cast<const unsigned int*>(
          &As[(wave * 16 + l15) * 32 + a_kmap(p, half)]);

    #pragma unroll
    for (int a = 0; a < 4; ++a) {
      FragU bfu;
      int n = a * 16 + l15;
      #pragma unroll
      for (int p = 0; p < 8; ++p)
        bfu.u[p] = *reinterpret_cast<const unsigned int*>(
            &Bs[n * 32 + a_kmap(p, half)]);
      acc[a] = __builtin_amdgcn_wmma_f32_16x16x32_bf16(
          false, af.b, false, bfu.b, (short)0, acc[a], false, false);
    }
    __syncthreads();
  }

  // epilogue
  #pragma unroll
  for (int a = 0; a < 4; ++a) {
    int gn = n0 + a * 16 + l15;
    float bv = bias ? bias[gn] : 0.0f;
    #pragma unroll
    for (int i = 0; i < 8; ++i) {
      int gm = m0 + wave * 16 + i + 8 * half;
      float val = acc[a][i] + bv;
      if (MODE == 1) {
        Cf[(size_t)gm * N + gn] = val;
      } else {
        int which = gn >> 10;        // 0=q 1=k 2=v
        int wn = gn & 1023;
        int h  = wn >> 6;
        int dd = wn & 63;
        int b  = gm >> 12;
        int t  = gm & 4095;
        size_t di = ((size_t)((b * H_CNT + h) * T_SEQ + t)) * HEAD_D + dd;
        unsigned short bfv = f2bf(val);
        if (which == 0)      Qo[di] = bfv;
        else if (which == 1) Ko[di] = bfv;
        else                 Vo[di] = bfv;
      }
    }
  }
}

// ---------------------------------------------------------------------------
// Per-(b,h,t) query squared-L2 norm (monotonic wrt. norm -> same top-k order)
// ---------------------------------------------------------------------------
__global__ void qnorm_kernel(const unsigned short* __restrict__ Qh,
                             float* __restrict__ norms) {
  int idx = blockIdx.x * blockDim.x + threadIdx.x;
  if (idx >= B_SZ * H_CNT * T_SEQ) return;
  const unsigned short* r = Qh + (size_t)idx * HEAD_D;
  float s = 0.0f;
  #pragma unroll 8
  for (int j = 0; j < HEAD_D; ++j) {
    float f = bf2f(r[j]);
    s += f * f;
  }
  norms[idx] = s;
}

// ---------------------------------------------------------------------------
// Top-KEEP per (b,h) via bitonic sort in LDS (4096 keys), then gather:
//   Ksel [bh][KEEP_PAD][64]  (row-major, tail rows zero)
//   Vt   [bh][64][KEEP_PAD]  (transposed so PV B-fragments are contiguous)
// ---------------------------------------------------------------------------
__global__ __launch_bounds__(1024) void topk_gather(
    const float* __restrict__ norms,
    const unsigned short* __restrict__ Kh,
    const unsigned short* __restrict__ Vh,
    unsigned short* __restrict__ Ksel,
    unsigned short* __restrict__ Vt) {
  __shared__ float key[T_SEQ];
  __shared__ int   sidx[T_SEQ];
  const int tid = threadIdx.x;
  const int bh  = blockIdx.x;
  for (int i = tid; i < T_SEQ; i += 1024) {
    key[i]  = norms[(size_t)bh * T_SEQ + i];
    sidx[i] = i;
  }
  __syncthreads();
  for (int k = 2; k <= T_SEQ; k <<= 1) {
    for (int j = k >> 1; j > 0; j >>= 1) {
      for (int i = tid; i < T_SEQ; i += 1024) {
        int p = i ^ j;
        if (p > i) {
          bool desc = ((i & k) == 0);
          float a = key[i], c = key[p];
          bool sw = desc ? (a < c) : (a > c);
          if (sw) {
            key[i] = c; key[p] = a;
            int t = sidx[i]; sidx[i] = sidx[p]; sidx[p] = t;
          }
        }
      }
      __syncthreads();
    }
  }
  unsigned short* Kb = Ksel + (size_t)bh * KEEP_PAD * HEAD_D;
  unsigned short* Vb = Vt   + (size_t)bh * HEAD_D * KEEP_PAD;
  for (int s = tid; s < KEEP_PAD; s += 1024) {
    uint4* kd = reinterpret_cast<uint4*>(Kb + (size_t)s * HEAD_D);
    if (s < KEEP_N) {
      int src = sidx[s];
      const uint4* kr = reinterpret_cast<const uint4*>(
          Kh + ((size_t)bh * T_SEQ + src) * HEAD_D);
      #pragma unroll
      for (int q = 0; q < 8; ++q) kd[q] = kr[q];
      const unsigned short* vr = Vh + ((size_t)bh * T_SEQ + src) * HEAD_D;
      #pragma unroll
      for (int dd = 0; dd < HEAD_D; ++dd)
        Vb[(size_t)dd * KEEP_PAD + s] = vr[dd];
    } else {
      uint4 z = make_uint4(0u, 0u, 0u, 0u);
      #pragma unroll
      for (int q = 0; q < 8; ++q) kd[q] = z;
      #pragma unroll
      for (int dd = 0; dd < HEAD_D; ++dd)
        Vb[(size_t)dd * KEEP_PAD + s] = 0;
    }
  }
}

// ---------------------------------------------------------------------------
// Flash-style sparse attention. Block = (b,h, 128-query tile), 8 waves; each
// wave owns 16 query rows. K/V 64-key chunks are staged in LDS once per block
// (coalesced b128), all waves build WMMA fragments from LDS. Branchless loop:
// buffers are zero-padded to KEEP_PAD, softmax mask is a v_cndmask select.
// ---------------------------------------------------------------------------
__global__ __launch_bounds__(256) void attn_wmma(
    const unsigned short* __restrict__ Qh,
    const unsigned short* __restrict__ Ksel,
    const unsigned short* __restrict__ Vt,
    unsigned short* __restrict__ Ctx) {
  __shared__ __align__(16) unsigned short Ks[64 * 64];   // [key][d]
  __shared__ __align__(16) unsigned short Vs[64 * 64];   // [d][key]
  __shared__ __align__(16) unsigned short Pl[8][16][72]; // per-wave P bounce
  const int tid  = threadIdx.x;
  const int wave = tid >> 5;
  const int lane = tid & 31;
  const int l15  = lane & 15;
  const int half = lane >> 4;
  const int bh = blockIdx.x >> 5;
  const int q0 = (blockIdx.x & 31) * 128;
  const int b  = bh >> 4;
  const int h  = bh & 15;

  // Q A-fragments (d = 64 -> two K-steps of 32), loaded once
  FragU qa[2];
  {
    const unsigned short* qrow =
        Qh + ((size_t)bh * T_SEQ + q0 + wave * 16 + l15) * HEAD_D;
    #pragma unroll
    for (int ks = 0; ks < 2; ++ks)
      #pragma unroll
      for (int p = 0; p < 8; ++p)
        qa[ks].u[p] = *reinterpret_cast<const unsigned int*>(
            qrow + ks * 32 + a_kmap(p, half));
  }

  float mi[8], rs[8];
  v8f O[4];
  #pragma unroll
  for (int i = 0; i < 8; ++i) { mi[i] = -3.0e38f; rs[i] = 0.0f; }
  #pragma unroll
  for (int a = 0; a < 4; ++a)
    #pragma unroll
    for (int i = 0; i < 8; ++i) O[a][i] = 0.0f;

  const unsigned short* Kb  = Ksel + (size_t)bh * KEEP_PAD * HEAD_D;
  const unsigned short* Vtb = Vt   + (size_t)bh * HEAD_D * KEEP_PAD;
  // staging coordinates (per thread): 64 rows x 64 cols, 16 bf16 per thread x2
  const int srow = tid >> 2;            // 0..63
  const int scol = (tid & 3) << 4;      // 0,16,32,48

  for (int kc = 0; kc < KEEP_N; kc += 64) {
    // ---- stage K chunk [key][d] and V chunk [d][key] into LDS ----
    {
      const uint4* ksrc = reinterpret_cast<const uint4*>(
          Kb + (size_t)(kc + srow) * HEAD_D + scol);
      uint4* kdst = reinterpret_cast<uint4*>(&Ks[srow * 64 + scol]);
      kdst[0] = ksrc[0];
      kdst[1] = ksrc[1];
      const uint4* vsrc = reinterpret_cast<const uint4*>(
          Vtb + (size_t)srow * KEEP_PAD + kc + scol);
      uint4* vdst = reinterpret_cast<uint4*>(&Vs[srow * 64 + scol]);
      vdst[0] = vsrc[0];
      vdst[1] = vsrc[1];
      if (kc + 64 < KEEP_PAD) {
        __builtin_prefetch(Kb + (size_t)(kc + 64 + srow) * HEAD_D + scol, 0, 1);
        __builtin_prefetch(Vtb + (size_t)srow * KEEP_PAD + kc + 64 + scol, 0, 1);
      }
    }
    __syncthreads();

    // ---- S = Q * Ksel^T (16 x 64 per wave) ----
    v8f S[4];
    #pragma unroll
    for (int a = 0; a < 4; ++a)
      #pragma unroll
      for (int i = 0; i < 8; ++i) S[a][i] = 0.0f;

    #pragma unroll
    for (int ks = 0; ks < 2; ++ks) {
      #pragma unroll
      for (int a = 0; a < 4; ++a) {
        FragU kf;
        #pragma unroll
        for (int p = 0; p < 8; ++p)
          kf.u[p] = *reinterpret_cast<const unsigned int*>(
              &Ks[(a * 16 + l15) * 64 + ks * 32 + a_kmap(p, half)]);
        S[a] = __builtin_amdgcn_wmma_f32_16x16x32_bf16(
            false, qa[ks].b, false, kf.b, (short)0, S[a], false, false);
      }
    }
    // ---- scale + tail mask (branchless select) ----
    #pragma unroll
    for (int a = 0; a < 4; ++a) {
      int keyc = kc + a * 16 + l15;
      #pragma unroll
      for (int i = 0; i < 8; ++i) {
        float s = S[a][i] * 0.125f;   // 1/sqrt(64)
        S[a][i] = (keyc < KEEP_N) ? s : -3.0e38f;
      }
    }
    // ---- chunk row-max (reduce across 16-lane half-group) ----
    float cm[8];
    #pragma unroll
    for (int i = 0; i < 8; ++i) {
      float x = fmaxf(fmaxf(S[0][i], S[1][i]), fmaxf(S[2][i], S[3][i]));
      x = fmaxf(x, __shfl_xor(x, 1));
      x = fmaxf(x, __shfl_xor(x, 2));
      x = fmaxf(x, __shfl_xor(x, 4));
      x = fmaxf(x, __shfl_xor(x, 8));
      cm[i] = x;
    }
    // ---- online rescale ----
    #pragma unroll
    for (int i = 0; i < 8; ++i) {
      float mnew  = fmaxf(mi[i], cm[i]);
      float alpha = __expf(mi[i] - mnew);
      rs[i] *= alpha;
      #pragma unroll
      for (int a = 0; a < 4; ++a) O[a][i] *= alpha;
      mi[i] = mnew;
    }
    // ---- P = exp(S - m), partial row sums, bounce to LDS ----
    #pragma unroll
    for (int a = 0; a < 4; ++a) {
      #pragma unroll
      for (int i = 0; i < 8; ++i) {
        float pv = __expf(S[a][i] - mi[i]);
        rs[i] += pv;
        Pl[wave][i + 8 * half][a * 16 + l15] = f2bf(pv);
      }
    }
    __syncthreads();
    // ---- re-read P as A-fragments; O += P * V ----
    FragU pa[2];
    #pragma unroll
    for (int ks = 0; ks < 2; ++ks)
      #pragma unroll
      for (int p = 0; p < 8; ++p)
        pa[ks].u[p] = *reinterpret_cast<const unsigned int*>(
            &Pl[wave][l15][ks * 32 + a_kmap(p, half)]);
    #pragma unroll
    for (int ks = 0; ks < 2; ++ks) {
      #pragma unroll
      for (int a = 0; a < 4; ++a) {
        FragU vf;
        #pragma unroll
        for (int p = 0; p < 8; ++p)
          vf.u[p] = *reinterpret_cast<const unsigned int*>(
              &Vs[(a * 16 + l15) * 64 + ks * 32 + a_kmap(p, half)]);
        O[a] = __builtin_amdgcn_wmma_f32_16x16x32_bf16(
            false, pa[ks].b, false, vf.b, (short)0, O[a], false, false);
      }
    }
    __syncthreads();
  }
  // ---- finalize: full row-sum, normalize, write ctx bf16 [B,T,D] ----
  #pragma unroll
  for (int i = 0; i < 8; ++i) {
    float x = rs[i];
    x += __shfl_xor(x, 1);
    x += __shfl_xor(x, 2);
    x += __shfl_xor(x, 4);
    x += __shfl_xor(x, 8);
    rs[i] = 1.0f / x;
  }
  #pragma unroll
  for (int a = 0; a < 4; ++a) {
    int dd = a * 16 + l15;
    #pragma unroll
    for (int i = 0; i < 8; ++i) {
      int t = q0 + wave * 16 + i + 8 * half;
      Ctx[((size_t)(b * T_SEQ + t)) * D_MODEL + h * HEAD_D + dd] =
          f2bf(O[a][i] * rs[i]);
    }
  }
}

// ---------------------------------------------------------------------------
extern "C" void kernel_launch(void* const* d_in, const int* in_sizes, int n_in,
                              void* d_out, int out_size, void* d_ws, size_t ws_size,
                              hipStream_t stream) {
  (void)in_sizes; (void)n_in; (void)out_size; (void)ws_size;
  const float* x      = (const float*)d_in[0];
  const float* W_qkv  = (const float*)d_in[1];
  const float* b_qkv  = (const float*)d_in[2];
  const float* W_proj = (const float*)d_in[3];
  const float* b_proj = (const float*)d_in[4];
  float* out = (float*)d_out;

  char* ws = (char*)d_ws;
  size_t off = 0;
  auto carve = [&](size_t bytes) -> void* {
    void* p = ws + off;
    off = (off + bytes + 255) & ~(size_t)255;
    return p;
  };
  unsigned short* xbf     = (unsigned short*)carve((size_t)M_TOT * D_MODEL * 2);
  unsigned short* wqkvbf  = (unsigned short*)carve((size_t)D_MODEL * N_QKV * 2);
  unsigned short* wprojbf = (unsigned short*)carve((size_t)D_MODEL * D_MODEL * 2);
  unsigned short* Qh = (unsigned short*)carve((size_t)B_SZ * H_CNT * T_SEQ * HEAD_D * 2);
  unsigned short* Kh = (unsigned short*)carve((size_t)B_SZ * H_CNT * T_SEQ * HEAD_D * 2);
  unsigned short* Vh = (unsigned short*)carve((size_t)B_SZ * H_CNT * T_SEQ * HEAD_D * 2);
  float* norms = (float*)carve((size_t)B_SZ * H_CNT * T_SEQ * 4);
  unsigned short* Ksel = (unsigned short*)carve((size_t)B_SZ * H_CNT * KEEP_PAD * HEAD_D * 2);
  unsigned short* Vt   = (unsigned short*)carve((size_t)B_SZ * H_CNT * HEAD_D * KEEP_PAD * 2);
  unsigned short* Ctx  = (unsigned short*)carve((size_t)M_TOT * D_MODEL * 2);

  // 1) precision conversion
  f32_to_bf16_kernel<<<2048, 256, 0, stream>>>(x, xbf, M_TOT * D_MODEL);
  f32_to_bf16_kernel<<<2048, 256, 0, stream>>>(W_qkv, wqkvbf, D_MODEL * N_QKV);
  f32_to_bf16_kernel<<<1024, 256, 0, stream>>>(W_proj, wprojbf, D_MODEL * D_MODEL);

  // 2) QKV projection (WMMA), scatter to head-major bf16 q/k/v
  dim3 g1(M_TOT / 128, N_QKV / 64);
  gemm_bf16_wmma<0><<<g1, 256, 0, stream>>>(
      xbf, wqkvbf, b_qkv, nullptr, Qh, Kh, Vh, M_TOT, N_QKV, D_MODEL);

  // 3) query norms + top-KEEP selection + K/V gather (K row-major, V transposed)
  qnorm_kernel<<<(B_SZ * H_CNT * T_SEQ + 255) / 256, 256, 0, stream>>>(Qh, norms);
  topk_gather<<<B_SZ * H_CNT, 1024, 0, stream>>>(norms, Kh, Vh, Ksel, Vt);

  // 4) flash-style sparse attention (WMMA, LDS-staged K/V chunks)
  attn_wmma<<<B_SZ * H_CNT * (T_SEQ / 128), 256, 0, stream>>>(Qh, Ksel, Vt, Ctx);

  // 5) output projection (WMMA), f32 to d_out
  dim3 g2(M_TOT / 128, D_MODEL / 64);
  gemm_bf16_wmma<1><<<g2, 256, 0, stream>>>(
      Ctx, wprojbf, b_proj, out, nullptr, nullptr, nullptr, M_TOT, D_MODEL, D_MODEL);
}
